// Attention_87162066305303
// MI455X (gfx1250) — matrix operand order repaired and checked by reading, not compiled, not run on hardware
//
#include <hip/hip_runtime.h>
#include <hip/hip_bf16.h>

// Problem constants
#define BB   2
#define SS   2048
#define HH   1024
#define NHD  16
#define HD   64
#define MTOT 4096   // B*S
#define N_QKV 3072
#define KDIM 1024

typedef __attribute__((ext_vector_type(16))) __bf16 v16bf;
typedef __attribute__((ext_vector_type(8)))  __bf16 v8bf;
typedef __attribute__((ext_vector_type(8)))  float  v8f;

__device__ __forceinline__ v8f vzero8() {
    v8f z;
#pragma unroll
    for (int i = 0; i < 8; ++i) z[i] = 0.0f;
    return z;
}

// D = A(16x32 bf16) * B(32x16 bf16) + C, f32 accumulate
__device__ __forceinline__ v8f wmma_bf16(v16bf a, v16bf b, v8f c) {
    return __builtin_amdgcn_wmma_f32_16x16x32_bf16(
        false, a, false, b, (short)0, c, false, false);
}

// A-fragment: 16x32 tile of row-major bf16 (row stride ld, elements).
// Lane L (m = L&15): half 0 holds K 0..15, half 1 holds K 16..31.
__device__ __forceinline__ v16bf load_a_frag(const __bf16* tile, int ld) {
    const int lane = threadIdx.x & 31;
    const int m  = lane & 15;
    const int kh = (lane >> 4) << 4;               // 0 or 16
    const __bf16* p = tile + (size_t)m * ld + kh;
    v8bf lo = *(const v8bf*)p;                     // K = kh .. kh+7
    v8bf hi = *(const v8bf*)(p + 8);               // K = kh+8 .. kh+15
    v16bf a;
#pragma unroll
    for (int i = 0; i < 8; ++i) { a[i] = lo[i]; a[8 + i] = hi[i]; }
    return a;
}

// B-fragment for Bt[k][n] = Wrows[n][k] (each lane reads a contiguous row).
// Lane L (n = L&15): half 0 holds K rows 0..15, half 1 holds K rows 16..31.
__device__ __forceinline__ v16bf load_bt_frag(const __bf16* wrows, int ld) {
    const int lane = threadIdx.x & 31;
    const int n  = lane & 15;
    const int kh = (lane >> 4) << 4;               // 0 or 16
    const __bf16* p = wrows + (size_t)n * ld + kh;
    v8bf lo = *(const v8bf*)p;                     // K = kh .. kh+7
    v8bf hi = *(const v8bf*)(p + 8);               // K = kh+8 .. kh+15
    v16bf b;
#pragma unroll
    for (int i = 0; i < 8; ++i) { b[i] = lo[i]; b[8 + i] = hi[i]; }
    return b;
}

// Async copy of one 16-byte chunk global -> LDS (ASYNCcnt-tracked).
__device__ __forceinline__ void async_copy_b128(const __bf16* gsrc, __bf16* ldst) {
    const unsigned int        lds = (unsigned int)(uintptr_t)ldst;  // low 32b = LDS offset
    const unsigned long long  g   = (unsigned long long)(uintptr_t)gsrc;
    asm volatile("global_load_async_to_lds_b128 %0, %1, off"
                 :: "v"(lds), "v"(g) : "memory");
}

__device__ __forceinline__ void wait_async_zero() {
    asm volatile("s_wait_asynccnt 0x0" ::: "memory");
}

// ---------------- fp32 -> bf16 conversion (8 elems/thread) ----------------
__global__ void cvt_f32_bf16_v8(const float* __restrict__ src,
                                __bf16* __restrict__ dst, int n8) {
    int i = blockIdx.x * blockDim.x + threadIdx.x;
    if (i < n8) {
        const float4 a = ((const float4*)src)[i * 2];
        const float4 b = ((const float4*)src)[i * 2 + 1];
        v8bf o;
        o[0] = (__bf16)a.x; o[1] = (__bf16)a.y; o[2] = (__bf16)a.z; o[3] = (__bf16)a.w;
        o[4] = (__bf16)b.x; o[5] = (__bf16)b.y; o[6] = (__bf16)b.z; o[7] = (__bf16)b.w;
        ((v8bf*)dst)[i] = o;
    }
}

// ---------------- QKV projection GEMM ----------------
// mixed[m, n] = sum_k X[m,k] * W[n,k] + bias[n]
// Epilogue scatters: n -> (h = n/192, t = (n%192)/64, d = n%64)
//   t==0: Q[bh, s, d] (pre-scaled by 1/8)   t==1: K[bh, s, d]   t==2: Vt[bh, d, s]
__device__ __forceinline__ void store_qkv_tile(
    const v8f& c, int m0, int n0, const float* __restrict__ bias,
    __bf16* __restrict__ Q, __bf16* __restrict__ K, __bf16* __restrict__ Vt)
{
    const int lane = threadIdx.x & 31;
    const int h   = n0 / 192;
    const int rem = n0 % 192;
    const int t   = rem / 64;
    const int d   = (rem % 64) + (lane & 15);
    const float bv = bias[n0 + (lane & 15)];
    const int rowoff = (lane >> 4) << 3;
#pragma unroll
    for (int e = 0; e < 8; ++e) {
        const int row = m0 + rowoff + e;           // 0..4095
        const int b   = row >> 11;                 // /2048
        const int s   = row & 2047;
        const float v = c[e] + bv;
        const size_t bh = (size_t)(b * NHD + h);
        if (t == 0)      Q[(bh * SS + s) * HD + d] = (__bf16)(v * 0.125f);
        else if (t == 1) K[(bh * SS + s) * HD + d] = (__bf16)v;
        else             Vt[(bh * HD + d) * SS + s] = (__bf16)v;
    }
}

__global__ __launch_bounds__(256) void qkv_gemm_kernel(
    const __bf16* __restrict__ X,    // [4096,1024]
    const __bf16* __restrict__ W,    // [3072,1024]
    const float*  __restrict__ bias, // [3072]
    __bf16* __restrict__ Q, __bf16* __restrict__ K, __bf16* __restrict__ Vt)
{
    const int wave = threadIdx.x >> 5;
    const int wy = wave >> 2, wx = wave & 3;
    const int mbase = blockIdx.y * 64  + wy * 32;
    const int nbase = blockIdx.x * 256 + wx * 64;

    v8f c[2][4];
#pragma unroll
    for (int i = 0; i < 2; ++i)
#pragma unroll
        for (int j = 0; j < 4; ++j) c[i][j] = vzero8();

    for (int k = 0; k < KDIM; k += 32) {
        v16bf a0 = load_a_frag(X + (size_t)mbase * KDIM + k, KDIM);
        v16bf a1 = load_a_frag(X + (size_t)(mbase + 16) * KDIM + k, KDIM);
        v16bf b[4];
#pragma unroll
        for (int j = 0; j < 4; ++j)
            b[j] = load_bt_frag(W + (size_t)(nbase + j * 16) * KDIM + k, KDIM);
#pragma unroll
        for (int j = 0; j < 4; ++j) {
            c[0][j] = wmma_bf16(a0, b[j], c[0][j]);
            c[1][j] = wmma_bf16(a1, b[j], c[1][j]);
        }
    }
#pragma unroll
    for (int i = 0; i < 2; ++i)
#pragma unroll
        for (int j = 0; j < 4; ++j)
            store_qkv_tile(c[i][j], mbase + i * 16, nbase + j * 16, bias, Q, K, Vt);
}

// ---------------- Flash attention (transposed formulation) ----------------
// Block = 8 waves sharing one (b,h); each wave owns a 16-query tile.
// K/V 32-key tiles are double-buffered in LDS: next tile's DMA
// (global_load_async_to_lds_b128, ASYNCcnt) overlaps current tile's WMMAs.
// S^T = K * Q^T (rows = keys, cols = queries), online softmax over rows,
// ctx^T = V^T * P^T, normalize, scatter ctx.
__global__ __launch_bounds__(256) void attn_kernel(
    const __bf16* __restrict__ Q,   // [32, 2048, 64]  (pre-scaled by 1/8)
    const __bf16* __restrict__ K,   // [32, 2048, 64]
    const __bf16* __restrict__ Vt,  // [32, 64, 2048]
    const float*  __restrict__ mask,// [2, 1, 2048, 2048]
    __bf16* __restrict__ Ctx)       // [4096, 1024]
{
    __shared__ __align__(16) __bf16 sK[2][32 * 64];  // keys kb..kb+31, ld 64
    __shared__ __align__(16) __bf16 sV[2][64 * 32];  // d rows 0..63,   ld 32

    const int tid   = threadIdx.x;                   // 0..255
    const int lane  = tid & 31;
    const int wave  = tid >> 5;                      // 0..7
    const int bh    = blockIdx.y;                    // 0..31
    const int b     = bh >> 4;
    const int h     = bh & 15;
    const int qbase = blockIdx.x * 128 + wave * 16;

    const __bf16* Qp = Q  + (size_t)bh * SS * HD;
    const __bf16* Kp = K  + (size_t)bh * SS * HD;
    const __bf16* Vp = Vt + (size_t)bh * HD * SS;
    const float*  Mp = mask + (size_t)b * SS * SS;

    // Q^T as B operand: frag j covers contraction d in [32j, 32j+32)
    const v16bf bq0 = load_bt_frag(Qp + (size_t)qbase * HD + 0,  HD);
    const v16bf bq1 = load_bt_frag(Qp + (size_t)qbase * HD + 32, HD);

    v8f acc[4];
#pragma unroll
    for (int i = 0; i < 4; ++i) acc[i] = vzero8();
    float m_run = -3.0e38f, l_run = 0.0f;

    const int q       = qbase + (lane & 15);   // this lane's query column
    const int rowoff  = (lane >> 4) << 3;      // C-layout M offset (0 or 8)
    const bool lowhalf = lane < 16;

    // Stage one 32-key step (K tile 4096B + V tile 4096B); 1 chunk each per thread.
    auto stage = [&](int buf, int kb) {
        // K tile: contiguous 4096B span; chunk tid -> sK[buf] + tid*8 elems
        async_copy_b128(Kp + (size_t)kb * HD + tid * 8, &sK[buf][tid * 8]);
        // V tile: 64 rows (stride SS) x 32 keys; chunk tid = d*4 + cc
        const int d  = tid >> 2;
        const int cc = tid & 3;
        async_copy_b128(Vp + (size_t)d * SS + kb + cc * 8,
                        &sV[buf][d * 32 + cc * 8]);
    };

    stage(0, 0);
    wait_async_zero();
    __syncthreads();

    for (int kb = 0; kb < SS; kb += 32) {
        const int cur = (kb >> 5) & 1;
        const bool more = (kb + 32) < SS;
        if (more) stage(cur ^ 1, kb + 32);          // DMA next tile, no wait
        if (more)
            __builtin_prefetch(Mp + (size_t)q * SS + kb + 32 + rowoff, 0, 1);

        // S^T tiles: st0 = keys kb..kb+15, st1 = keys kb+16..kb+31 (cols = queries)
        const __bf16* cK = &sK[cur][0];
        const __bf16* cV = &sV[cur][0];
        v8f st0 = vzero8(), st1 = vzero8();
        st0 = wmma_bf16(load_a_frag(cK,                64), bq0, st0);
        st0 = wmma_bf16(load_a_frag(cK + 32,           64), bq1, st0);
        st1 = wmma_bf16(load_a_frag(cK + 16 * 64,      64), bq0, st1);
        st1 = wmma_bf16(load_a_frag(cK + 16 * 64 + 32, 64), bq1, st1);

        // mask: scores*mask - 10000*(1-mask); lane holds contiguous key runs
        const v8f mk0 = *(const v8f*)(Mp + (size_t)q * SS + kb + rowoff);
        const v8f mk1 = *(const v8f*)(Mp + (size_t)q * SS + kb + 16 + rowoff);
#pragma unroll
        for (int e = 0; e < 8; ++e) {
            st0[e] = st0[e] * mk0[e] - 10000.0f * (1.0f - mk0[e]);
            st1[e] = st1[e] * mk1[e] - 10000.0f * (1.0f - mk1[e]);
        }

        // column (per-query) max across 32 keys: 16 in-lane + lane^16 partner
        float mx = st0[0];
#pragma unroll
        for (int e = 1; e < 8; ++e) mx = fmaxf(mx, st0[e]);
#pragma unroll
        for (int e = 0; e < 8; ++e) mx = fmaxf(mx, st1[e]);
        mx = fmaxf(mx, __shfl_xor(mx, 16));
        const float m_new = fmaxf(m_run, mx);

        float p0[8], p1[8], sum = 0.0f;
#pragma unroll
        for (int e = 0; e < 8; ++e) {
            p0[e] = __expf(st0[e] - m_new);
            p1[e] = __expf(st1[e] - m_new);
            sum += p0[e] + p1[e];
        }
        sum += __shfl_xor(sum, 16);
        const float alpha = __expf(m_run - m_new);
        l_run = l_run * alpha + sum;
        m_run = m_new;
#pragma unroll
        for (int i = 0; i < 4; ++i)
#pragma unroll
            for (int e = 0; e < 8; ++e) acc[i][e] *= alpha;

        // Build P^T B-fragment: lane half0 needs keys kb+0..15, half1 keys kb+16..31
        v16bf pb;
#pragma unroll
        for (int e = 0; e < 8; ++e) {
            const float o0 = __shfl_xor(p0[e], 16);
            const float o1 = __shfl_xor(p1[e], 16);
            pb[e]     = (__bf16)(lowhalf ? p0[e] : o1);
            pb[8 + e] = (__bf16)(lowhalf ? o0 : p1[e]);
        }

        // ctx^T += V^T * P^T  (4 d-tiles of 16)
#pragma unroll
        for (int dt = 0; dt < 4; ++dt)
            acc[dt] = wmma_bf16(load_a_frag(cV + dt * 16 * 32, 32), pb, acc[dt]);

        if (more) wait_async_zero();   // next tile landed (own wave's DMAs)
        __syncthreads();               // everyone done reading cur + staging done
    }

    const float inv = 1.0f / l_run;
    const size_t outbase = ((size_t)b * SS + q) * HH + h * HD;
#pragma unroll
    for (int dt = 0; dt < 4; ++dt)
#pragma unroll
        for (int e = 0; e < 8; ++e) {
            const int d = dt * 16 + rowoff + e;
            Ctx[outbase + d] = (__bf16)(acc[dt][e] * inv);
        }
}

// ---------------- Dense output GEMM ----------------
__global__ __launch_bounds__(256) void dense_gemm_kernel(
    const __bf16* __restrict__ Ctx,  // [4096,1024]
    const __bf16* __restrict__ W,    // [1024,1024]
    const float*  __restrict__ bias, // [1024]
    float* __restrict__ out)         // [4096,1024] f32
{
    const int lane = threadIdx.x & 31;
    const int wave = threadIdx.x >> 5;
    const int wy = wave >> 2, wx = wave & 3;
    const int mbase = blockIdx.y * 64  + wy * 32;
    const int nbase = blockIdx.x * 256 + wx * 64;

    v8f c[2][4];
#pragma unroll
    for (int i = 0; i < 2; ++i)
#pragma unroll
        for (int j = 0; j < 4; ++j) c[i][j] = vzero8();

    for (int k = 0; k < KDIM; k += 32) {
        v16bf a0 = load_a_frag(Ctx + (size_t)mbase * KDIM + k, KDIM);
        v16bf a1 = load_a_frag(Ctx + (size_t)(mbase + 16) * KDIM + k, KDIM);
        v16bf b[4];
#pragma unroll
        for (int j = 0; j < 4; ++j)
            b[j] = load_bt_frag(W + (size_t)(nbase + j * 16) * KDIM + k, KDIM);
#pragma unroll
        for (int j = 0; j < 4; ++j) {
            c[0][j] = wmma_bf16(a0, b[j], c[0][j]);
            c[1][j] = wmma_bf16(a1, b[j], c[1][j]);
        }
    }

    const int rowoff = (lane >> 4) << 3;
#pragma unroll
    for (int mi = 0; mi < 2; ++mi)
#pragma unroll
        for (int ni = 0; ni < 4; ++ni) {
            const int n  = nbase + ni * 16 + (lane & 15);
            const float bv = bias[n];
#pragma unroll
            for (int e = 0; e < 8; ++e) {
                const int row = mbase + mi * 16 + rowoff + e;
                out[(size_t)row * HH + n] = c[mi][ni][e] + bv;
            }
        }
}

// ---------------- launch ----------------
extern "C" void kernel_launch(void* const* d_in, const int* in_sizes, int n_in,
                              void* d_out, int out_size, void* d_ws, size_t ws_size,
                              hipStream_t stream) {
    (void)in_sizes; (void)n_in; (void)out_size; (void)ws_size;
    const float* hidden  = (const float*)d_in[0];
    const float* mask    = (const float*)d_in[1];
    const float* qkv_w   = (const float*)d_in[2];
    const float* qkv_b   = (const float*)d_in[3];
    const float* dense_w = (const float*)d_in[4];
    const float* dense_b = (const float*)d_in[5];
    float* out = (float*)d_out;

    char* ws = (char*)d_ws;
    __bf16* Xb  = (__bf16*)(ws + 0);          // 4096x1024   (8 MB)
    __bf16* Wq  = (__bf16*)(ws + 8388608);    // 3072x1024   (6 MB)
    __bf16* Wd  = (__bf16*)(ws + 14680064);   // 1024x1024   (2 MB)
    __bf16* Q   = (__bf16*)(ws + 16777216);   // 32x2048x64  (8 MB)
    __bf16* K   = (__bf16*)(ws + 25165824);   // 32x2048x64  (8 MB)
    __bf16* Vt  = (__bf16*)(ws + 33554432);   // 32x64x2048  (8 MB)
    __bf16* Ctx = (__bf16*)(ws + 41943040);   // 4096x1024   (8 MB)

    cvt_f32_bf16_v8<<<(MTOT * KDIM / 8 + 255) / 256, 256, 0, stream>>>(hidden, Xb, MTOT * KDIM / 8);
    cvt_f32_bf16_v8<<<(N_QKV * KDIM / 8 + 255) / 256, 256, 0, stream>>>(qkv_w, Wq, N_QKV * KDIM / 8);
    cvt_f32_bf16_v8<<<(HH * KDIM / 8 + 255) / 256, 256, 0, stream>>>(dense_w, Wd, HH * KDIM / 8);

    qkv_gemm_kernel<<<dim3(N_QKV / 256, MTOT / 64), 256, 0, stream>>>(Xb, Wq, qkv_b, Q, K, Vt);
    attn_kernel<<<dim3(SS / 128, BB * NHD), 256, 0, stream>>>(Q, K, Vt, mask, Ctx);
    dense_gemm_kernel<<<dim3(HH / 256, MTOT / 64), 256, 0, stream>>>(Ctx, Wd, dense_b, out);
}
